// SelfAttention_28106265985236
// MI455X (gfx1250) — matrix operand order repaired and checked by reading, not compiled, not run on hardware
//
#include <hip/hip_runtime.h>
#include <hip/hip_bf16.h>
#include <math.h>

#define DIM  1024
#define NSEQ 4096

typedef __attribute__((ext_vector_type(16))) __bf16 v16bf;
typedef __attribute__((ext_vector_type(8)))  __bf16 v8bf;
typedef __attribute__((ext_vector_type(4)))  float  v4f;
typedef __attribute__((ext_vector_type(8)))  float  v8f;
typedef __attribute__((ext_vector_type(4)))  unsigned int v4u;
typedef __attribute__((ext_vector_type(4)))  int    v4i;
typedef __attribute__((ext_vector_type(8)))  int    v8i;

#if defined(__AMDGCN__) && __has_builtin(__builtin_amdgcn_tensor_load_to_lds) && \
    __has_builtin(__builtin_amdgcn_s_wait_tensorcnt)
#define HAVE_TDM 1
#else
#define HAVE_TDM 0
#endif

// LDS partition for the attention kernel (dynamic shared)
#define K_LDS_OFF   0                       // 32 x 1024 bf16 = 65536 B
#define V_LDS_OFF   65536                   // 1024 x 32 bf16 = 65536 B
#define S_LDS_OFF   131072                  // 16 x 32 f32    = 2048 B
#define P_LDS_OFF   133120                  // 16 x 32 bf16   = 1024 B
#define M_LDS_OFF   134144                  // 16 f32
#define L_LDS_OFF   134208                  // 16 f32
#define A_LDS_OFF   134272                  // 16 f32
#define SMEM_BYTES  134336

#if HAVE_TDM
// ---- Tensor DMA descriptor builders (ISA 8.3 / 8.4, 2-D tensor, 2B elems) ----
__device__ inline v4u tdm_g0(unsigned lds_addr, unsigned long long gaddr) {
    v4u g;
    g[0] = 1u;                                   // count=1, is_restore=0, no gather
    g[1] = lds_addr;                             // lds_addr[63:32]
    g[2] = (unsigned)gaddr;                      // global_addr[95:64]
    g[3] = (unsigned)((gaddr >> 32) & 0x1FFFFFFull) | (2u << 30); // addr hi | type=2
    return g;
}
__device__ inline v8i tdm_g1(unsigned tdim0, unsigned tdim1,
                             unsigned tile0, unsigned tile1, unsigned stride0) {
    v8i g;
    g[0] = (int)(1u << 16);                      // wg_mask=0, data_size=1 (2 bytes)
    g[1] = (int)((tdim0 & 0xFFFFu) << 16);       // abar=0 | tensor_dim0 lo16
    g[2] = (int)((tdim0 >> 16) | ((tdim1 & 0xFFFFu) << 16)); // d0 hi | d1 lo
    g[3] = (int)((tdim1 >> 16) | (tile0 << 16)); // d1 hi | tile_dim0
    g[4] = (int)tile1;                           // tile_dim1 | tile_dim2=0
    g[5] = (int)stride0;                         // tensor_dim0_stride lo32
    g[6] = 0;                                    // stride0 hi | dim1_stride lo
    g[7] = 0;
    return g;
}
__device__ inline void tdm_load(v4u g0, v8i g1) {
    v4i z4 = {0, 0, 0, 0};
#if __clang_major__ >= 23
    v8i z8 = {0, 0, 0, 0, 0, 0, 0, 0};
    __builtin_amdgcn_tensor_load_to_lds(g0, g1, z4, z4, z8, 0);
#else
    __builtin_amdgcn_tensor_load_to_lds(g0, g1, z4, z4, 0);
#endif
}
#endif

// ---------------------------------------------------------------------------
// Kernel 0a: Xb = bf16(X), fully coalesced, 8 elements/thread.
// ---------------------------------------------------------------------------
__global__ __launch_bounds__(256) void convert_x(
    const float* __restrict__ X, __bf16* __restrict__ Xb)
{
    const size_t idx = ((size_t)blockIdx.x * 256 + threadIdx.x) * 8;
    v4f f0 = *(const v4f*)(X + idx);
    v4f f1 = *(const v4f*)(X + idx + 4);
    v8bf o;
    #pragma unroll
    for (int i = 0; i < 4; ++i) { o[i] = (__bf16)f0[i]; o[4 + i] = (__bf16)f1[i]; }
    *(v8bf*)(Xb + idx) = o;
}

// ---------------------------------------------------------------------------
// Kernel 0b: Wt[n][k] = bf16(W[k][n]) * scale (transpose; Q gets 1/sqrt(d)).
// ---------------------------------------------------------------------------
__global__ __launch_bounds__(256) void transpose_w(
    const float* __restrict__ Wq, const float* __restrict__ Wk,
    const float* __restrict__ Wv, __bf16* __restrict__ Wtq,
    __bf16* __restrict__ Wtk, __bf16* __restrict__ Wtv)
{
    const int mat = blockIdx.y;
    const float* W = (mat == 0) ? Wq : (mat == 1 ? Wk : Wv);
    __bf16*    Wt  = (mat == 0) ? Wtq : (mat == 1 ? Wtk : Wtv);
    const float scale = (mat == 0) ? 0.03125f : 1.0f;

    const int n0 = (blockIdx.x & 15) * 64;
    const int k0 = (blockIdx.x >> 4) * 64;
    const int n  = n0 + (threadIdx.x & 63);
    const int kb = k0 + (threadIdx.x >> 6) * 16;

    v8bf o0, o1;
    #pragma unroll
    for (int i = 0; i < 8; ++i) {
        o0[i] = (__bf16)(W[(size_t)(kb + i) * DIM + n] * scale);
        o1[i] = (__bf16)(W[(size_t)(kb + 8 + i) * DIM + n] * scale);
    }
    *(v8bf*)(Wt + (size_t)n * DIM + kb)     = o0;
    *(v8bf*)(Wt + (size_t)n * DIM + kb + 8) = o1;
}

// ---------------------------------------------------------------------------
// Kernel 1: Y = Xb @ Wt^T. 32x32 output per wave (2x2 WMMA tiles), all
// fragment loads contiguous b128. V output stored transposed (Vt[n][j]).
// ---------------------------------------------------------------------------
__global__ __launch_bounds__(256) void qkv_gemm(
    const __bf16* __restrict__ Xb,
    const __bf16* __restrict__ Wtq, const __bf16* __restrict__ Wtk,
    const __bf16* __restrict__ Wtv,
    __bf16* __restrict__ Qb, __bf16* __restrict__ Kb, __bf16* __restrict__ Vt)
{
    const int lane = threadIdx.x & 31;
    const int wave = threadIdx.x >> 5;
    const int wg   = blockIdx.x * 8 + wave;

    const int TILES = (NSEQ / 32) * (DIM / 32); // 4096 per matrix
    const int mat = wg / TILES;
    const int rem = wg % TILES;
    const int m0  = (rem / (DIM / 32)) * 32;
    const int n0  = (rem % (DIM / 32)) * 32;

    const __bf16* Wt = (mat == 0) ? Wtq : (mat == 1 ? Wtk : Wtv);
    const int row = lane & 15;
    const int hi  = lane >> 4;

    v8f acc[2][2];
    #pragma unroll
    for (int i = 0; i < 2; ++i)
        #pragma unroll
        for (int j = 0; j < 2; ++j) acc[i][j] = (v8f){};

    for (int k0 = 0; k0 < DIM; k0 += 32) {
        v16bf a[2], b[2];
        #pragma unroll
        for (int t = 0; t < 2; ++t) {
            const __bf16* ap = Xb + (size_t)(m0 + t * 16 + row) * DIM + k0 + 8 * hi;
            v8bf g0 = *(const v8bf*)ap;
            v8bf g1 = *(const v8bf*)(ap + 16);
            const __bf16* bp = Wt + (size_t)(n0 + t * 16 + row) * DIM + k0 + 8 * hi;
            v8bf h0 = *(const v8bf*)bp;
            v8bf h1 = *(const v8bf*)(bp + 16);
            #pragma unroll
            for (int i = 0; i < 8; ++i) {
                a[t][i] = g0[i]; a[t][8 + i] = g1[i];
                b[t][i] = h0[i]; b[t][8 + i] = h1[i];
            }
        }
        #pragma unroll
        for (int i = 0; i < 2; ++i)
            #pragma unroll
            for (int j = 0; j < 2; ++j)
                acc[i][j] = __builtin_amdgcn_wmma_f32_16x16x32_bf16(
                    false, a[i], false, b[j], (short)0, acc[i][j], false, false);
    }

    #pragma unroll
    for (int ti = 0; ti < 2; ++ti) {
        #pragma unroll
        for (int tj = 0; tj < 2; ++tj) {
            const int rbase = m0 + ti * 16 + 8 * hi;
            const int c     = n0 + tj * 16 + row;
            if (mat == 2) {
                v8bf o;
                #pragma unroll
                for (int i = 0; i < 8; ++i) o[i] = (__bf16)acc[ti][tj][i];
                *(v8bf*)(Vt + (size_t)c * NSEQ + rbase) = o;
            } else {
                __bf16* Y = (mat == 0) ? Qb : Kb;
                #pragma unroll
                for (int i = 0; i < 8; ++i)
                    Y[(size_t)(rbase + i) * DIM + c] = (__bf16)acc[ti][tj][i];
            }
        }
    }
}

// ---------------------------------------------------------------------------
// Kernel 2: fused flash attention. Block = 16 query rows, 8 waves.
// TDM stages the 32-key K chunk (32x1024) and V chunk (1024x32 of Vt) into
// LDS; waves compute S (d-split + ds_add_f32 reduce), online softmax, P@V.
// ---------------------------------------------------------------------------
__global__ __launch_bounds__(256) void flash_attn_bf16(
    const __bf16* __restrict__ Qb,
    const __bf16* __restrict__ Kb,
    const __bf16* __restrict__ Vt,
    float* __restrict__ Out)
{
    extern __shared__ char smem[];
    __bf16* Kld = (__bf16*)(smem + K_LDS_OFF);
    __bf16* Vld = (__bf16*)(smem + V_LDS_OFF);
    float*  Sld = (float*)(smem + S_LDS_OFF);
    __bf16* Pld = (__bf16*)(smem + P_LDS_OFF);
    float*  mrow = (float*)(smem + M_LDS_OFF);
    float*  lrow = (float*)(smem + L_LDS_OFF);
    float*  arow = (float*)(smem + A_LDS_OFF);

    const int tid  = threadIdx.x;
    const int lane = tid & 31;
    const int wave = tid >> 5;
    const int q0   = blockIdx.x * 16;
    const int row  = lane & 15;
    const int hi   = lane >> 4;

    if (tid < 16) { mrow[tid] = -INFINITY; lrow[tid] = 0.0f; }

    // Hoisted Q fragments: this wave covers d-range [wave*128, wave*128+128)
    v16bf qf[4];
    #pragma unroll
    for (int kk = 0; kk < 4; ++kk) {
        const int k0 = wave * 128 + kk * 32;
        const __bf16* qp = Qb + (size_t)(q0 + row) * DIM + k0 + 8 * hi;
        v8bf g0 = *(const v8bf*)qp;
        v8bf g1 = *(const v8bf*)(qp + 16);
        #pragma unroll
        for (int i = 0; i < 8; ++i) { qf[kk][i] = g0[i]; qf[kk][8 + i] = g1[i]; }
    }

    v8f acc[8];
    #pragma unroll
    for (int t = 0; t < 8; ++t) acc[t] = (v8f){};

    __syncthreads();

    for (int j0 = 0; j0 < NSEQ; j0 += 32) {
#if HAVE_TDM
        if (wave == 0) {
            // K chunk: rows j0..j0+31, all 1024 cols -> Kld[32][1024]
            tdm_load(
                tdm_g0((unsigned)(size_t)Kld,
                       (unsigned long long)(size_t)(Kb + (size_t)j0 * DIM)),
                tdm_g1(DIM, NSEQ, DIM, 32, DIM));
            // V chunk: all 1024 rows of Vt, cols j0..j0+31 -> Vld[1024][32]
            tdm_load(
                tdm_g0((unsigned)(size_t)Vld,
                       (unsigned long long)(size_t)(Vt + j0)),
                tdm_g1(NSEQ, DIM, 32, DIM, NSEQ));
            __builtin_amdgcn_s_wait_tensorcnt((short)0);
        }
#else
        if (j0 + 32 < NSEQ) {
            __builtin_prefetch(Kb + (size_t)(j0 + 32 + row) * DIM + wave * 128, 0, 0);
            __builtin_prefetch(Vt + (size_t)(wave * 128 + row) * NSEQ + j0 + 32, 0, 0);
        }
#endif
        Sld[tid] = 0.0f;
        Sld[tid + 256] = 0.0f;
        __syncthreads();   // publishes TDM-filled LDS + zeroed S to all waves

        // ---- partial S = Q[:, dslice] @ K[:, dslice]^T ----
        v8f s0 = {}, s1 = {};
        #pragma unroll
        for (int kk = 0; kk < 4; ++kk) {
            const int k0 = wave * 128 + kk * 32;
#if HAVE_TDM
            const __bf16* kp0 = Kld + (size_t)row * DIM + k0 + 8 * hi;
            const __bf16* kp1 = Kld + (size_t)(row + 16) * DIM + k0 + 8 * hi;
#else
            const __bf16* kp0 = Kb + (size_t)(j0 + row) * DIM + k0 + 8 * hi;
            const __bf16* kp1 = Kb + (size_t)(j0 + 16 + row) * DIM + k0 + 8 * hi;
#endif
            v8bf h0 = *(const v8bf*)kp0;
            v8bf h1 = *(const v8bf*)(kp0 + 16);
            v8bf h2 = *(const v8bf*)kp1;
            v8bf h3 = *(const v8bf*)(kp1 + 16);
            v16bf b0, b1;
            #pragma unroll
            for (int i = 0; i < 8; ++i) {
                b0[i] = h0[i]; b0[8 + i] = h1[i];
                b1[i] = h2[i]; b1[8 + i] = h3[i];
            }
            s0 = __builtin_amdgcn_wmma_f32_16x16x32_bf16(
                false, qf[kk], false, b0, (short)0, s0, false, false);
            s1 = __builtin_amdgcn_wmma_f32_16x16x32_bf16(
                false, qf[kk], false, b1, (short)0, s1, false, false);
        }
        #pragma unroll
        for (int i = 0; i < 8; ++i) {
            const int m = i + 8 * hi;
            atomicAdd(&Sld[m * 32 + row],      s0[i]);
            atomicAdd(&Sld[m * 32 + 16 + row], s1[i]);
        }
        __syncthreads();

        // ---- online softmax: one lane per query row ----
        if (tid < 16) {
            const int r = tid;
            const float mo = mrow[r];
            float mx = mo;
            for (int j = 0; j < 32; ++j) mx = fmaxf(mx, Sld[r * 32 + j]);
            const float alpha = __expf(mo - mx);
            float sum = 0.0f;
            for (int j = 0; j < 32; ++j) {
                const float p = __expf(Sld[r * 32 + j] - mx);
                sum += p;
                Pld[r * 32 + j] = (__bf16)p;
            }
            mrow[r] = mx;
            lrow[r] = lrow[r] * alpha + sum;
            arow[r] = alpha;
        }
        __syncthreads();

        // ---- rescale running O, then O += P @ V ----
        float a8[8];
        #pragma unroll
        for (int i = 0; i < 8; ++i) a8[i] = arow[i + 8 * hi];
        #pragma unroll
        for (int t = 0; t < 8; ++t)
            #pragma unroll
            for (int i = 0; i < 8; ++i) acc[t][i] *= a8[i];

        v16bf pf;
        {
            const int kb = 8 * hi;
            v8bf p0 = *(const v8bf*)&Pld[row * 32 + kb];
            v8bf p1 = *(const v8bf*)&Pld[row * 32 + 16 + kb];
            #pragma unroll
            for (int i = 0; i < 8; ++i) { pf[i] = p0[i]; pf[8 + i] = p1[i]; }
        }
        #pragma unroll
        for (int t = 0; t < 8; ++t) {
            const int n0 = wave * 128 + t * 16;
#if HAVE_TDM
            const __bf16* vp = Vld + (size_t)(n0 + row) * 32 + 8 * hi;
#else
            const __bf16* vp = Vt + (size_t)(n0 + row) * NSEQ + j0 + 8 * hi;
#endif
            v8bf h0 = *(const v8bf*)vp;
            v8bf h1 = *(const v8bf*)(vp + 16);
            v16bf vf;
            #pragma unroll
            for (int i = 0; i < 8; ++i) { vf[i] = h0[i]; vf[8 + i] = h1[i]; }
            acc[t] = __builtin_amdgcn_wmma_f32_16x16x32_bf16(
                false, pf, false, vf, (short)0, acc[t], false, false);
        }
        __syncthreads();
    }

    float l8[8];
    #pragma unroll
    for (int i = 0; i < 8; ++i) l8[i] = lrow[i + 8 * hi];
    #pragma unroll
    for (int t = 0; t < 8; ++t) {
        const int n0 = wave * 128 + t * 16;
        #pragma unroll
        for (int i = 0; i < 8; ++i)
            Out[(size_t)(q0 + i + 8 * hi) * DIM + n0 + row] = acc[t][i] / l8[i];
    }
}

// ---------------------------------------------------------------------------
extern "C" void kernel_launch(void* const* d_in, const int* in_sizes, int n_in,
                              void* d_out, int out_size, void* d_ws, size_t ws_size,
                              hipStream_t stream) {
    (void)in_sizes; (void)n_in; (void)out_size; (void)ws_size;

    const float* X  = (const float*)d_in[0];
    const float* Wq = (const float*)d_in[1];
    const float* Wk = (const float*)d_in[2];
    const float* Wv = (const float*)d_in[3];
    float* Out = (float*)d_out;

    const size_t ND = (size_t)NSEQ * DIM; // 4M
    const size_t DD = (size_t)DIM * DIM;  // 1M
    __bf16* Xb  = (__bf16*)d_ws;
    __bf16* Wtq = Xb + ND;
    __bf16* Wtk = Wtq + DD;
    __bf16* Wtv = Wtk + DD;
    __bf16* Qb  = Wtv + DD;
    __bf16* Kb  = Qb + ND;
    __bf16* Vt  = Kb + ND;   // stored transposed [DIM][NSEQ]

    convert_x<<<(int)(ND / (256 * 8)), 256, 0, stream>>>(X, Xb);
    dim3 tg((DIM / 64) * (DIM / 64), 3, 1);
    transpose_w<<<tg, 256, 0, stream>>>(Wq, Wk, Wv, Wtq, Wtk, Wtv);

    const int total_tiles = 3 * (NSEQ / 32) * (DIM / 32); // 12288 waves
    qkv_gemm<<<total_tiles / 8, 256, 0, stream>>>(Xb, Wtq, Wtk, Wtv, Qb, Kb, Vt);
    flash_attn_bf16<<<NSEQ / 16, 256, SMEM_BYTES, stream>>>(Qb, Kb, Vt, Out);
}
